// Induction_66374424592574
// MI455X (gfx1250) — matrix operand order repaired and checked by reading, not compiled, not run on hardware
//
#include <hip/hip_runtime.h>
#include <hip/hip_bf16.h>

#define HH 512
#define CC 64
#define SS 2048
#define MM (CC * SS)   // 131072 rows

typedef __attribute__((ext_vector_type(16))) __bf16 v16bf;
typedef __attribute__((ext_vector_type(8)))  __bf16 v8bf;
typedef __attribute__((ext_vector_type(8)))  float  v8f;

union V16 { v16bf v; v8bf h[2]; };

__device__ __forceinline__ v16bf ld_frag(const __bf16* p0, const __bf16* p1) {
  V16 u;
  u.h[0] = *(const v8bf*)p0;   // 16B aligned by construction
  u.h[1] = *(const v8bf*)p1;
  return u.v;
}

// ---------------------------------------------------------------------------
// Prep: split W into bf16 hi/lo (bf16x3 trick), zero routing logits b.
// ---------------------------------------------------------------------------
__global__ __launch_bounds__(256) void prep_kernel(const float* __restrict__ W,
                                                   unsigned short* __restrict__ Wh_,
                                                   unsigned short* __restrict__ Wl_,
                                                   float* __restrict__ b) {
  const int i = blockIdx.x * 256 + threadIdx.x;
  if (i < HH * HH) {
    float v = W[i];
    __bf16 h = (__bf16)v;
    __bf16 l = (__bf16)(v - (float)h);
    ((__bf16*)Wh_)[i] = h;
    ((__bf16*)Wl_)[i] = l;
  }
  if (i < MM) b[i] = 0.0f;
}

// ---------------------------------------------------------------------------
// Fused GEMM + bias + squash:  e[m][n] = squash_row( sum_k x[m][k]*W[n][k] + bias[n] )
// BM=64 rows/block, BN=512 (full row -> fused squash), BK=32, 16 waves.
// bf16x3: acc += Ah*Bh + Al*Bh + Ah*Bl   (fp32 accumulate in v_wmma)
// ---------------------------------------------------------------------------
__global__ __launch_bounds__(512) void gemm_squash_kernel(
    const float* __restrict__ x,
    const unsigned short* __restrict__ Wh_,
    const unsigned short* __restrict__ Wl_,
    const float* __restrict__ bias,
    float* __restrict__ e) {
  const __bf16* Wh = (const __bf16*)Wh_;
  const __bf16* Wl = (const __bf16*)Wl_;
  constexpr int BM = 64, BK = 32, LDA = BK + 8;   // stride 40 halves = 80B (16B-aligned, conflict-free)

  __shared__ __align__(16) __bf16 As_h[BM * LDA];
  __shared__ __align__(16) __bf16 As_l[BM * LDA];
  __shared__ float row_sq[BM];

  const int tid   = threadIdx.x;
  const int wave  = tid >> 5;
  const int lane  = tid & 31;
  const int mg    = wave >> 2;       // 0..3 : 16-row group
  const int nq    = wave & 3;        // 0..3 : 128-col group
  const int lrow  = lane & 15;
  const int khalf = lane >> 4;       // 0/1
  const size_t m0 = (size_t)blockIdx.x * BM;

  // staging coords: one float4 per thread (2048 floats = 512 float4)
  const int srow = tid >> 3;         // 0..63
  const int skc  = (tid & 7) * 4;    // 0..28

  if (tid < BM) row_sq[tid] = 0.0f;

  v8f acc[8];
#pragma unroll
  for (int t = 0; t < 8; ++t) { v8f z = {}; acc[t] = z; }

  for (int k0 = 0; k0 < HH; k0 += BK) {
    __syncthreads();
    // stage A tile (64x32 fp32 -> hi/lo bf16): 1x b128 load + 2x b64 LDS stores/thread
    {
      const float4 v4 = *(const float4*)&x[(m0 + (size_t)srow) * HH + k0 + skc];
      const float vv[4] = {v4.x, v4.y, v4.z, v4.w};
      union { __bf16 b[4]; unsigned long long u; } ph, pl;
#pragma unroll
      for (int q = 0; q < 4; ++q) {
        const __bf16 h = (__bf16)vv[q];
        ph.b[q] = h;
        pl.b[q] = (__bf16)(vv[q] - (float)h);
      }
      *(unsigned long long*)&As_h[srow * LDA + skc] = ph.u;
      *(unsigned long long*)&As_l[srow * LDA + skc] = pl.u;
    }
    __syncthreads();

    // A fragments (contiguous k order used here, same as B: K = i + (lane>=16)*16)
    const int arow = mg * 16 + lrow;
    const int ka   = khalf * 16;
    const v16bf a_h = ld_frag(&As_h[arow * LDA + ka], &As_h[arow * LDA + ka + 8]);
    const v16bf a_l = ld_frag(&As_l[arow * LDA + ka], &As_l[arow * LDA + ka + 8]);

#pragma unroll
    for (int t = 0; t < 8; ++t) {
      // B fragment (32x16: K = i + (lane>=16)*16, col = lane&15); W is L2-resident
      const int n = nq * 128 + t * 16 + lrow;
      const size_t boff = (size_t)n * HH + k0 + khalf * 16;
      const v16bf b_h = ld_frag(&Wh[boff], &Wh[boff + 8]);
      const v16bf b_l = ld_frag(&Wl[boff], &Wl[boff + 8]);
      acc[t] = __builtin_amdgcn_wmma_f32_16x16x32_bf16(false, a_h, false, b_h, (short)0, acc[t], false, false);
      acc[t] = __builtin_amdgcn_wmma_f32_16x16x32_bf16(false, a_l, false, b_h, (short)0, acc[t], false, false);
      acc[t] = __builtin_amdgcn_wmma_f32_16x16x32_bf16(false, a_h, false, b_l, (short)0, acc[t], false, false);
    }
  }

  // bias + per-row sum of squares (C/D layout: elem r -> m = (lane>=16)*8 + r, n = lane&15)
  float sq[8];
#pragma unroll
  for (int r = 0; r < 8; ++r) sq[r] = 0.0f;
#pragma unroll
  for (int t = 0; t < 8; ++t) {
    const int n = nq * 128 + t * 16 + lrow;
    const float bv = bias[n];
#pragma unroll
    for (int r = 0; r < 8; ++r) {
      float v = acc[t][r] + bv;
      acc[t][r] = v;
      sq[r] += v * v;
    }
  }
  // reduce over the 16 lanes of each half-wave (xor masks 1,2,4,8 stay in-half)
#pragma unroll
  for (int r = 0; r < 8; ++r) {
#pragma unroll
    for (int m = 1; m < 16; m <<= 1) sq[r] += __shfl_xor(sq[r], m, 32);
  }
  const int moff = mg * 16 + khalf * 8;
  if (lrow == 0) {
#pragma unroll
    for (int r = 0; r < 8; ++r) atomicAdd(&row_sq[moff + r], sq[r]);
  }
  __syncthreads();

  float scl[8];
#pragma unroll
  for (int r = 0; r < 8; ++r) {
    const float s2 = row_sq[moff + r];
    scl[r] = (s2 / (1.0f + s2)) * rsqrtf(s2 + 1e-7f);
  }
#pragma unroll
  for (int t = 0; t < 8; ++t) {
    const int n = nq * 128 + t * 16 + lrow;
#pragma unroll
    for (int r = 0; r < 8; ++r)
      e[(m0 + (size_t)(moff + r)) * HH + n] = acc[t][r] * scl[r];
  }
}

// ---------------------------------------------------------------------------
// Routing iteration kernels (memory-bound streaming over e)
// ---------------------------------------------------------------------------
__global__ __launch_bounds__(256) void softmax_kernel(const float* __restrict__ b,
                                                      float* __restrict__ d,
                                                      float* __restrict__ c_raw) {
  const int c = blockIdx.x;
  const int tid = threadIdx.x;
  __shared__ float red[256];
  // zero the accumulator for this iteration's weighted sum
  c_raw[c * HH + tid]       = 0.0f;
  c_raw[c * HH + tid + 256] = 0.0f;

  float vals[8];
  float mx = -3.402823466e38f;
#pragma unroll
  for (int i = 0; i < 8; ++i) {
    vals[i] = b[c * SS + tid + i * 256];
    mx = fmaxf(mx, vals[i]);
  }
  red[tid] = mx;
  __syncthreads();
  for (int s = 128; s > 0; s >>= 1) {
    if (tid < s) red[tid] = fmaxf(red[tid], red[tid + s]);
    __syncthreads();
  }
  mx = red[0];
  __syncthreads();
  float sum = 0.0f;
#pragma unroll
  for (int i = 0; i < 8; ++i) { vals[i] = expf(vals[i] - mx); sum += vals[i]; }
  red[tid] = sum;
  __syncthreads();
  for (int s = 128; s > 0; s >>= 1) {
    if (tid < s) red[tid] += red[tid + s];
    __syncthreads();
  }
  const float inv = 1.0f / red[0];
#pragma unroll
  for (int i = 0; i < 8; ++i) d[c * SS + tid + i * 256] = vals[i] * inv;
}

__global__ __launch_bounds__(256) void wsum_kernel(const float* __restrict__ e,
                                                   const float* __restrict__ d,
                                                   float* __restrict__ c_raw) {
  const int c = blockIdx.x;
  const int j = blockIdx.y;            // 16-way split over S
  const int tid = threadIdx.x;
  const float* ec = e + ((size_t)c * SS + (size_t)j * 128) * HH;
  const float* dc = d + c * SS + j * 128;
  float a0 = 0.0f, a1 = 0.0f;
#pragma unroll 4
  for (int s = 0; s < 128; ++s) {
    if ((s & 7) == 0) {  // stream-ahead prefetch of the e rows (global_prefetch_b8)
      __builtin_prefetch(&ec[(size_t)(s + 16) * HH + tid], 0, 0);
      __builtin_prefetch(&ec[(size_t)(s + 16) * HH + tid + 256], 0, 0);
    }
    const float w = dc[s];
    a0 += w * ec[(size_t)s * HH + tid];
    a1 += w * ec[(size_t)s * HH + tid + 256];
  }
  atomicAdd(&c_raw[c * HH + tid], a0);
  atomicAdd(&c_raw[c * HH + tid + 256], a1);
}

__global__ __launch_bounds__(256) void squashc_kernel(const float* __restrict__ c_raw,
                                                      float* __restrict__ c_out) {
  const int c = blockIdx.x;
  const int tid = threadIdx.x;
  __shared__ float red[256];
  const float v0 = c_raw[c * HH + tid];
  const float v1 = c_raw[c * HH + tid + 256];
  red[tid] = v0 * v0 + v1 * v1;
  __syncthreads();
  for (int s = 128; s > 0; s >>= 1) {
    if (tid < s) red[tid] += red[tid + s];
    __syncthreads();
  }
  const float s2 = red[0];
  const float scl = (s2 / (1.0f + s2)) * rsqrtf(s2 + 1e-7f);
  c_out[c * HH + tid]       = v0 * scl;
  c_out[c * HH + tid + 256] = v1 * scl;
}

// b[c][s] += dot(e[c][s][:], c_i[c][:])  — one wave32 per row, 16 floats/lane
__global__ __launch_bounds__(256) void bupd_kernel(const float* __restrict__ e,
                                                   const float* __restrict__ ci,
                                                   float* __restrict__ b) {
  const int row  = blockIdx.x * 8 + (threadIdx.x >> 5);
  const int lane = threadIdx.x & 31;
  const int c    = row >> 11;   // / SS
  const float4* er = (const float4*)(e + (size_t)row * HH) + lane * 4;
  const float4* cr = (const float4*)(ci + (size_t)c * HH) + lane * 4;
  float s = 0.0f;
#pragma unroll
  for (int q = 0; q < 4; ++q) {
    const float4 a = er[q];
    const float4 w = cr[q];
    s += a.x * w.x + a.y * w.y + a.z * w.z + a.w * w.w;
  }
#pragma unroll
  for (int m = 1; m < 32; m <<= 1) s += __shfl_xor(s, m, 32);
  if (lane == 0) b[row] += s;
}

// ---------------------------------------------------------------------------
extern "C" void kernel_launch(void* const* d_in, const int* in_sizes, int n_in,
                              void* d_out, int out_size, void* d_ws, size_t ws_size,
                              hipStream_t stream) {
  (void)in_sizes; (void)n_in; (void)out_size; (void)ws_size;
  const float* x    = (const float*)d_in[0];
  const float* W    = (const float*)d_in[1];
  const float* bias = (const float*)d_in[2];
  float* out = (float*)d_out;

  char* ws = (char*)d_ws;
  size_t off = 0;
  float* e = (float*)(ws + off);           off += (size_t)MM * HH * sizeof(float);  // 268 MB
  unsigned short* Wh = (unsigned short*)(ws + off); off += (size_t)HH * HH * 2;
  unsigned short* Wl = (unsigned short*)(ws + off); off += (size_t)HH * HH * 2;
  float* bij   = (float*)(ws + off);       off += (size_t)MM * sizeof(float);
  float* dij   = (float*)(ws + off);       off += (size_t)MM * sizeof(float);
  float* c_raw = (float*)(ws + off);       off += (size_t)CC * HH * sizeof(float);

  prep_kernel<<<(HH * HH + 255) / 256, 256, 0, stream>>>(W, Wh, Wl, bij);
  gemm_squash_kernel<<<MM / 64, 512, 0, stream>>>(x, Wh, Wl, bias, e);
  for (int it = 0; it < 3; ++it) {
    softmax_kernel<<<CC, 256, 0, stream>>>(bij, dij, c_raw);
    wsum_kernel<<<dim3(CC, 16), 256, 0, stream>>>(e, dij, c_raw);
    squashc_kernel<<<CC, 256, 0, stream>>>(c_raw, out);
    if (it < 2) bupd_kernel<<<MM / 8, 256, 0, stream>>>(e, out, bij);  // last update unused
  }
}
